// TransformerLayer_2972117369059
// MI455X (gfx1250) — matrix operand order repaired and checked by reading, not compiled, not run on hardware
//
#include <hip/hip_runtime.h>

// ---------------------------------------------------------------------------
// Transformer encoder layer for MI455X (gfx1250, wave32, WMMA + TDM).
// B=4, S=1024, D=768, H=12, FF=3072.  All GEMMs are C = A * Bt^T with A,Bt
// row-major bf16, f32 accumulation via v_wmma_f32_16x16x32_bf16.
// LDS tiles are filled by the Tensor Data Mover (tensor_load_to_lds,
// TENSORcnt-tracked, double-buffered) when the builtin is available.
// Workspace requirement: ~560 MB (bump-allocated from d_ws).
// ---------------------------------------------------------------------------

typedef __bf16 bf16_t;
typedef __attribute__((ext_vector_type(16))) __bf16 v16bf;
typedef __attribute__((ext_vector_type(8)))  __bf16 v8bf;
typedef __attribute__((ext_vector_type(8)))  float  v8f;

typedef unsigned int u32x4 __attribute__((ext_vector_type(4)));
typedef int          i32x8 __attribute__((ext_vector_type(8)));
typedef int          i32x4 __attribute__((ext_vector_type(4)));

#define Bb 4
#define Ss 1024
#define Dd 768
#define Hh 12
#define FFf 3072

#if defined(__has_builtin)
#if __has_builtin(__builtin_amdgcn_tensor_load_to_lds)
#define USE_TDM 1
#endif
#endif
#ifndef USE_TDM
#define USE_TDM 0
#endif

#if defined(__has_builtin)
#if __has_builtin(__builtin_amdgcn_s_wait_tensorcnt)
#define TDM_WAIT0() __builtin_amdgcn_s_wait_tensorcnt(0)
#endif
#endif
#ifndef TDM_WAIT0
#define TDM_WAIT0() asm volatile("s_wait_tensorcnt 0x0" ::: "memory")
#endif

#if USE_TDM
// ---------------------------------------------------------------------------
// Issue one TDM 2D tile load: rows x cols bf16 elements, source row stride
// `ld` elements, destination LDS rows padded by 16B after each 128B row
// (matches bf16_t[rows][72] when cols == 64).
// D# layout per CDNA5 ISA 8.3/8.4.
// ---------------------------------------------------------------------------
__device__ inline void tdm_load_tile(const bf16_t* gsrc, void* ldst,
                                     int rows, int cols, int ld) {
  const unsigned long long ga = (unsigned long long)gsrc;
  const unsigned lds = (unsigned)(unsigned long long)ldst;  // low 32b = LDS offset
  u32x4 g0;
  g0.x = 1u;                                            // count=1, user descriptor
  g0.y = lds;                                           // lds_addr
  g0.z = (unsigned)(ga & 0xFFFFFFFFu);                  // global_addr[31:0]
  g0.w = (unsigned)((ga >> 32) & 0x1FFFFFFu)            // global_addr[56:32]
         | (2u << 30);                                  // type=2 (image)
  i32x8 g1;
  // d0: wg_mask=0 | data_size=1(2B)<<16 | pad_enable<<20
  //     | pad_interval=4 (32 DWORDs = one 128B row) << 22
  //     | pad_amount = 3 (4 DWORDs = 16B) << 25
  g1[0] = (1 << 16) | (1 << 20) | (4 << 22) | (3 << 25);
  g1[1] = (ld & 0xFFFF) << 16;                          // tensor_dim0[15:0]
  g1[2] = (ld >> 16) & 0xFFFF;                          // tensor_dim0[31:16] | tdim1 lo16 (0)
  g1[3] = 0x10 | ((cols & 0xFFFF) << 16);               // tensor_dim1 = 1<<20 | tile_dim0
  g1[4] = rows & 0xFFFF;                                // tile_dim1 | tile_dim2=0
  g1[5] = ld;                                           // tensor_dim0_stride[31:0]
  g1[6] = 0;                                            // stride hi | dim1_stride lo
  g1[7] = 0;                                            // dim1_stride hi
  i32x4 g2 = {0, 0, 0, 0};
  i32x4 g3 = {0, 0, 0, 0};
#if __clang_major__ >= 23
  i32x8 g4 = {0, 0, 0, 0, 0, 0, 0, 0};
  __builtin_amdgcn_tensor_load_to_lds(g0, g1, g2, g3, g4, 0);
#else
  __builtin_amdgcn_tensor_load_to_lds(g0, g1, g2, g3, 0);
#endif
}
#endif  // USE_TDM

// ---------------------------------------------------------------------------
// f32 -> bf16 elementwise convert
// ---------------------------------------------------------------------------
__global__ void k_cvt_bf16(const float* __restrict__ in, bf16_t* __restrict__ out, long n) {
  long i = (long)blockIdx.x * blockDim.x + threadIdx.x;
  long stride = (long)gridDim.x * blockDim.x;
  for (; i < n; i += stride) out[i] = (bf16_t)in[i];
}

// ---------------------------------------------------------------------------
// f32 [R,C] -> bf16 [C,R] transposed convert (batched over blockIdx.z)
// ---------------------------------------------------------------------------
__global__ __launch_bounds__(256) void k_cvt_transpose(
    const float* __restrict__ in, bf16_t* __restrict__ out, int R, int C) {
  __shared__ float tile[32][33];
  const long zoff = (long)blockIdx.z * (long)R * (long)C;
  in  += zoff;
  out += zoff;
  const int c0 = blockIdx.x * 32, r0 = blockIdx.y * 32;
  const int tx = threadIdx.x & 31, ty = threadIdx.x >> 5;  // 32 x 8
  for (int i = ty; i < 32; i += 8)
    tile[i][tx] = in[(long)(r0 + i) * C + (c0 + tx)];
  __syncthreads();
  for (int i = ty; i < 32; i += 8)
    out[(long)(c0 + i) * R + (r0 + tx)] = (bf16_t)tile[tx][i];
}

// ---------------------------------------------------------------------------
// WMMA GEMM:  C[M,N] = act( scale * (A[M,K] * Bt[N,K]^T) + bias[N] )
//   Block tile 128x128, K-step 64; 8 waves (2x4), 4x2 accumulators per wave.
//   TDM path: wave 0 DMAs next K tiles into the alternate LDS buffer while
//   all waves run WMMA on the current one (one barrier per K-step).
//   Requires M%128==0, N%128==0, K%64==0 (true for all uses here).
// ---------------------------------------------------------------------------
#if USE_TDM
#define NBUF 2
#else
#define NBUF 1
#endif

template <int RELU, int TSTORE, typename OutT>
__global__ __launch_bounds__(256) void k_gemm_bt(
    const bf16_t* __restrict__ A, const bf16_t* __restrict__ Bt,
    OutT* __restrict__ C, const float* __restrict__ bias,
    int M, int N, int K, int lda, int ldb, int ldc,
    long a_sh, long a_sb, long b_sh, long b_sb,
    long c_sh, long c_sb, long bias_sh, int ZB, float scale) {
  (void)M; (void)N;
  const int z = blockIdx.z;
  const int zh = z / ZB, zb = z % ZB;
  A  += zh * a_sh + zb * a_sb;
  Bt += zh * b_sh + zb * b_sb;
  C  += zh * c_sh + zb * c_sb;
  if (bias) bias += zh * bias_sh;

  const int m0 = blockIdx.y * 128;
  const int n0 = blockIdx.x * 128;

  __shared__ bf16_t As[NBUF][128][72];   // 128 x (64 + 8 pad) bf16 = 18 KB each
  __shared__ bf16_t Bs[NBUF][128][72];

  const int t    = threadIdx.x;
  const int wave = t >> 5;
  const int lane = t & 31;
  const int wm   = wave >> 2;        // 0..1 -> 64-row slab
  const int wn   = wave & 3;         // 0..3 -> 32-col slab
  const int half = lane >> 4;        // 0: lanes 0-15, 1: lanes 16-31
  const int l16  = lane & 15;

  v8f acc[4][2];
#pragma unroll
  for (int i = 0; i < 4; ++i)
#pragma unroll
    for (int j = 0; j < 2; ++j)
#pragma unroll
      for (int r = 0; r < 8; ++r) acc[i][j][r] = 0.0f;

#if USE_TDM
  // prologue: DMA first K tiles into buffer 0
  if (wave == 0) {
    tdm_load_tile(A  + (long)m0 * lda, &As[0][0][0], 128, 64, lda);
    tdm_load_tile(Bt + (long)n0 * ldb, &Bs[0][0][0], 128, 64, ldb);
  }
#else
  const int lr = t >> 3;             // 0..31
  const int lc = (t & 7) * 8;        // 0..56 step 8 (bf16 elems)
#endif

  int buf = 0;
  for (int k0 = 0; k0 < K; k0 += 64) {
#if USE_TDM
    TDM_WAIT0();                     // wave 0: current tiles landed in LDS
    __syncthreads();                 // release all waves; prior compute done
    if (wave == 0 && k0 + 64 < K) {  // DMA next tiles into alternate buffer
      tdm_load_tile(A  + (long)m0 * lda + (k0 + 64), &As[buf ^ 1][0][0], 128, 64, lda);
      tdm_load_tile(Bt + (long)n0 * ldb + (k0 + 64), &Bs[buf ^ 1][0][0], 128, 64, ldb);
    }
#else
    __syncthreads();
#pragma unroll
    for (int p = 0; p < 4; ++p) {
      const int r = p * 32 + lr;
      *(v8bf*)&As[0][r][lc] = *(const v8bf*)(A  + (long)(m0 + r) * lda + k0 + lc);
      *(v8bf*)&Bs[0][r][lc] = *(const v8bf*)(Bt + (long)(n0 + r) * ldb + k0 + lc);
    }
    if (k0 + 64 < K) {  // hint next K tile toward L2 (global_prefetch_b8)
      __builtin_prefetch(A  + (long)(m0 + lr) * lda + k0 + 64 + lc, 0, 1);
      __builtin_prefetch(Bt + (long)(n0 + lr) * ldb + k0 + 64 + lc, 0, 1);
    }
    __syncthreads();
#endif

#pragma unroll
    for (int kk = 0; kk < 64; kk += 32) {
      // A fragment (16x32): lanes 0-15 hold K {0..7,16..23}, lanes 16-31 {8..15,24..31}
      v16bf af[4];
      const int ao = kk + (half ? 8 : 0);
#pragma unroll
      for (int i = 0; i < 4; ++i) {
        const bf16_t* pa = &As[buf][wm * 64 + i * 16 + l16][ao];
        v8bf lo = *(const v8bf*)pa;
        v8bf hi = *(const v8bf*)(pa + 16);
        af[i] = __builtin_shufflevector(lo, hi, 0, 1, 2, 3, 4, 5, 6, 7,
                                        8, 9, 10, 11, 12, 13, 14, 15);
      }
      // B fragment (32x16 = rows of Bt): lanes 0-15 hold K {0..7,16..23}, lanes 16-31 {8..15,24..31}
      v16bf bfm[2];
      const int bo = kk + (half ? 8 : 0);
#pragma unroll
      for (int j = 0; j < 2; ++j) {
        const bf16_t* pb = &Bs[buf][wn * 32 + j * 16 + l16][bo];
        v8bf lo = *(const v8bf*)pb;
        v8bf hi = *(const v8bf*)(pb + 16);
        bfm[j] = __builtin_shufflevector(lo, hi, 0, 1, 2, 3, 4, 5, 6, 7,
                                         8, 9, 10, 11, 12, 13, 14, 15);
      }
#pragma unroll
      for (int i = 0; i < 4; ++i)
#pragma unroll
        for (int j = 0; j < 2; ++j)
          acc[i][j] = __builtin_amdgcn_wmma_f32_16x16x32_bf16(
              false, af[i], false, bfm[j], (short)0, acc[i][j], false, false);
    }
#if USE_TDM
    buf ^= 1;
#endif
  }

  // epilogue: C/D layout — reg r, lane l: row = r + (half?8:0), col = l16
  const int mbase = m0 + wm * 64;
  const int nbase = n0 + wn * 32;
#pragma unroll
  for (int i = 0; i < 4; ++i) {
#pragma unroll
    for (int j = 0; j < 2; ++j) {
      const int n = nbase + j * 16 + l16;
      const float bv = bias ? bias[n] : 0.0f;
#pragma unroll
      for (int r = 0; r < 8; ++r) {
        const int m = mbase + i * 16 + (half ? 8 + r : r);
        float v = acc[i][j][r] * scale + bv;
        if (RELU) v = fmaxf(v, 0.0f);
        if (TSTORE) C[(long)n * ldc + m] = (OutT)v;
        else        C[(long)m * ldc + n] = (OutT)v;
      }
    }
  }
}

// ---------------------------------------------------------------------------
// In-place stable softmax over rows of length 1024 (bf16 storage, f32 math).
// ---------------------------------------------------------------------------
__device__ inline float wave_max(float v) {
#pragma unroll
  for (int o = 16; o; o >>= 1) v = fmaxf(v, __shfl_xor(v, o, 32));
  return v;
}
__device__ inline float wave_sum(float v) {
#pragma unroll
  for (int o = 16; o; o >>= 1) v += __shfl_xor(v, o, 32);
  return v;
}

__global__ __launch_bounds__(256) void k_softmax_1024(bf16_t* __restrict__ p) {
  __shared__ float red[8];
  p += (long)blockIdx.x * 1024;
  const int t = threadIdx.x, wave = t >> 5, lane = t & 31;

  float vals[4], mx = -3.0e38f;
#pragma unroll
  for (int i = 0; i < 4; ++i) {
    vals[i] = (float)p[t + i * 256];
    mx = fmaxf(mx, vals[i]);
  }
  mx = wave_max(mx);
  if (lane == 0) red[wave] = mx;
  __syncthreads();
  mx = red[0];
#pragma unroll
  for (int w = 1; w < 8; ++w) mx = fmaxf(mx, red[w]);
  __syncthreads();

  float sum = 0.0f;
#pragma unroll
  for (int i = 0; i < 4; ++i) {
    vals[i] = __expf(vals[i] - mx);
    sum += vals[i];
  }
  sum = wave_sum(sum);
  if (lane == 0) red[wave] = sum;
  __syncthreads();
  sum = 0.0f;
#pragma unroll
  for (int w = 0; w < 8; ++w) sum += red[w];
  const float inv = 1.0f / sum;
#pragma unroll
  for (int i = 0; i < 4; ++i) p[t + i * 256] = (bf16_t)(vals[i] * inv);
}

// ---------------------------------------------------------------------------
// out = LayerNorm(x + y) * g + be over rows of 768 (f32); optional bf16 copy.
// ---------------------------------------------------------------------------
__global__ __launch_bounds__(256) void k_add_ln_768(
    const float* __restrict__ x, const float* __restrict__ y,
    const float* __restrict__ g, const float* __restrict__ be,
    float* __restrict__ out_f32, bf16_t* __restrict__ out_b16) {
  __shared__ float red[8];
  const long row = blockIdx.x;
  const int t = threadIdx.x, wave = t >> 5, lane = t & 31;
  x += row * Dd; y += row * Dd;

  float v[3], sum = 0.0f;
#pragma unroll
  for (int i = 0; i < 3; ++i) {
    v[i] = x[t + i * 256] + y[t + i * 256];
    sum += v[i];
  }
  sum = wave_sum(sum);
  if (lane == 0) red[wave] = sum;
  __syncthreads();
  sum = 0.0f;
#pragma unroll
  for (int w = 0; w < 8; ++w) sum += red[w];
  const float mu = sum * (1.0f / Dd);
  __syncthreads();

  float vs = 0.0f;
#pragma unroll
  for (int i = 0; i < 3; ++i) {
    const float d = v[i] - mu;
    vs += d * d;
  }
  vs = wave_sum(vs);
  if (lane == 0) red[wave] = vs;
  __syncthreads();
  vs = 0.0f;
#pragma unroll
  for (int w = 0; w < 8; ++w) vs += red[w];
  const float rs = rsqrtf(vs * (1.0f / Dd) + 1.0e-6f);

#pragma unroll
  for (int i = 0; i < 3; ++i) {
    const int c = t + i * 256;
    const float o = (v[i] - mu) * rs * g[c] + be[c];
    if (out_f32) out_f32[row * Dd + c] = o;
    if (out_b16) out_b16[row * Dd + c] = (bf16_t)o;
  }
}

// ---------------------------------------------------------------------------
// host-side orchestration
// ---------------------------------------------------------------------------
extern "C" void kernel_launch(void* const* d_in, const int* in_sizes, int n_in,
                              void* d_out, int out_size, void* d_ws, size_t ws_size,
                              hipStream_t stream) {
  (void)in_sizes; (void)n_in; (void)out_size; (void)ws_size;
  const float* x   = (const float*)d_in[0];
  const float* Wq  = (const float*)d_in[1];
  const float* bq  = (const float*)d_in[2];
  const float* Wk  = (const float*)d_in[3];
  const float* bk  = (const float*)d_in[4];
  const float* Wv  = (const float*)d_in[5];
  const float* bv  = (const float*)d_in[6];
  const float* Wo  = (const float*)d_in[7];
  const float* bo  = (const float*)d_in[8];
  const float* W1  = (const float*)d_in[9];
  const float* b1  = (const float*)d_in[10];
  const float* W2  = (const float*)d_in[11];
  const float* b2  = (const float*)d_in[12];
  const float* g1  = (const float*)d_in[13];
  const float* be1 = (const float*)d_in[14];
  const float* g2  = (const float*)d_in[15];
  const float* be2 = (const float*)d_in[16];

  const long T = (long)Bb * Ss;          // 4096 tokens
  char* wsp = (char*)d_ws;
  auto alloc = [&](size_t bytes) -> void* {
    void* p = (void*)wsp;
    wsp += (bytes + 255) & ~(size_t)255;
    return p;
  };

  bf16_t* xb     = (bf16_t*)alloc(T * Dd * 2);                       // x bf16
  bf16_t* WqT    = (bf16_t*)alloc((size_t)Hh * Dd * Dd * 2);
  bf16_t* WkT    = (bf16_t*)alloc((size_t)Hh * Dd * Dd * 2);
  bf16_t* WvT    = (bf16_t*)alloc((size_t)Hh * Dd * Dd * 2);
  bf16_t* WoT    = (bf16_t*)alloc((size_t)Dd * Hh * Dd * 2);         // [768,9216]
  bf16_t* W1T    = (bf16_t*)alloc((size_t)FFf * Dd * 2);             // [3072,768]
  bf16_t* W2T    = (bf16_t*)alloc((size_t)Dd * FFf * 2);             // [768,3072]
  bf16_t* qb     = (bf16_t*)alloc((size_t)Hh * T * Dd * 2);          // [H,4096,768]
  bf16_t* kb     = (bf16_t*)alloc((size_t)Hh * T * Dd * 2);
  bf16_t* vTb    = (bf16_t*)alloc((size_t)Hh * Dd * T * 2);          // [H,768,4096]
  bf16_t* att    = (bf16_t*)alloc((size_t)Hh * Bb * Ss * Ss * 2);    // [H,B,S,S]
  bf16_t* concat = (bf16_t*)alloc((size_t)T * Hh * Dd * 2);          // [4096,9216]
  float*  mha_f  = (float*)alloc((size_t)T * Dd * 4);
  float*  h_f    = (float*)alloc((size_t)T * Dd * 4);
  bf16_t* hb     = (bf16_t*)alloc((size_t)T * Dd * 2);
  bf16_t* ffb    = (bf16_t*)alloc((size_t)T * FFf * 2);
  float*  ff2_f  = (float*)alloc((size_t)T * Dd * 4);

  const float sscale = 1.0f / sqrtf((float)Dd);

  // ---- precision staging: x and transposed weights to bf16 ----------------
  k_cvt_bf16<<<3072, 256, 0, stream>>>(x, xb, T * Dd);
  k_cvt_transpose<<<dim3(Dd / 32, Dd / 32, Hh), 256, 0, stream>>>(Wq, WqT, Dd, Dd);
  k_cvt_transpose<<<dim3(Dd / 32, Dd / 32, Hh), 256, 0, stream>>>(Wk, WkT, Dd, Dd);
  k_cvt_transpose<<<dim3(Dd / 32, Dd / 32, Hh), 256, 0, stream>>>(Wv, WvT, Dd, Dd);
  k_cvt_transpose<<<dim3(Dd / 32, (Hh * Dd) / 32, 1), 256, 0, stream>>>(Wo, WoT, Hh * Dd, Dd);
  k_cvt_transpose<<<dim3(FFf / 32, Dd / 32, 1), 256, 0, stream>>>(W1, W1T, Dd, FFf);
  k_cvt_transpose<<<dim3(Dd / 32, FFf / 32, 1), 256, 0, stream>>>(W2, W2T, FFf, Dd);

  // ---- Q, K, V projections (batched over heads) ---------------------------
  k_gemm_bt<0, 0, bf16_t><<<dim3(6, 32, Hh), 256, 0, stream>>>(
      xb, WqT, qb, bq, (int)T, Dd, Dd, Dd, Dd, Dd,
      0L, 0L, (long)Dd * Dd, 0L, (long)T * Dd, 0L, (long)Dd, 1, 1.0f);
  k_gemm_bt<0, 0, bf16_t><<<dim3(6, 32, Hh), 256, 0, stream>>>(
      xb, WkT, kb, bk, (int)T, Dd, Dd, Dd, Dd, Dd,
      0L, 0L, (long)Dd * Dd, 0L, (long)T * Dd, 0L, (long)Dd, 1, 1.0f);
  // v stored transposed per head: vT[h][e][t]  ([768,4096])
  k_gemm_bt<0, 1, bf16_t><<<dim3(6, 32, Hh), 256, 0, stream>>>(
      xb, WvT, vTb, bv, (int)T, Dd, Dd, Dd, Dd, (int)T,
      0L, 0L, (long)Dd * Dd, 0L, (long)Dd * T, 0L, (long)Dd, 1, 1.0f);

  // ---- scores = scale * q . k^T   per (h,b), z = h*4+b --------------------
  k_gemm_bt<0, 0, bf16_t><<<dim3(8, 8, Hh * Bb), 256, 0, stream>>>(
      qb, kb, att, nullptr, Ss, Ss, Dd, Dd, Dd, Ss,
      (long)T * Dd, (long)Ss * Dd,         // A (q) strides over h, b
      (long)T * Dd, (long)Ss * Dd,         // Bt (k)
      (long)Bb * Ss * Ss, (long)Ss * Ss,   // C (scores [H,B,S,S])
      0L, Bb, sscale);

  // ---- softmax over last dim ----------------------------------------------
  k_softmax_1024<<<Hh * Bb * Ss, 256, 0, stream>>>(att);

  // ---- heads = attn . v = attn . vT^T ; write directly into concat layout -
  k_gemm_bt<0, 0, bf16_t><<<dim3(6, 8, Hh * Bb), 256, 0, stream>>>(
      att, vTb, concat, nullptr, Ss, Dd, Ss, Ss, (int)T, Hh * Dd,
      (long)Bb * Ss * Ss, (long)Ss * Ss,   // A (attn)
      (long)Dd * T, (long)Ss,              // Bt (vT): h stride, b col offset
      (long)Dd, (long)Ss * Hh * Dd,        // C (concat [4096, 9216])
      0L, Bb, 1.0f);

  // ---- mha = concat @ Wo + bo  (f32 out) ----------------------------------
  k_gemm_bt<0, 0, float><<<dim3(6, 32, 1), 256, 0, stream>>>(
      concat, WoT, mha_f, bo, (int)T, Dd, Hh * Dd, Hh * Dd, Hh * Dd, Dd,
      0L, 0L, 0L, 0L, 0L, 0L, 0L, 1, 1.0f);

  // ---- h = LN(x + mha) ----------------------------------------------------
  k_add_ln_768<<<(int)T, 256, 0, stream>>>(x, mha_f, g1, be1, h_f, hb);

  // ---- ff = relu(h @ W1 + b1) ---------------------------------------------
  k_gemm_bt<1, 0, bf16_t><<<dim3(24, 32, 1), 256, 0, stream>>>(
      hb, W1T, ffb, b1, (int)T, FFf, Dd, Dd, Dd, FFf,
      0L, 0L, 0L, 0L, 0L, 0L, 0L, 1, 1.0f);

  // ---- ff2 = ff @ W2 + b2  (f32 out) --------------------------------------
  k_gemm_bt<0, 0, float><<<dim3(6, 32, 1), 256, 0, stream>>>(
      ffb, W2T, ff2_f, b2, (int)T, Dd, FFf, FFf, FFf, Dd,
      0L, 0L, 0L, 0L, 0L, 0L, 0L, 1, 1.0f);

  // ---- out = LN(h + ff2) --------------------------------------------------
  k_add_ln_768<<<(int)T, 256, 0, stream>>>(h_f, ff2_f, g2, be2, (float*)d_out, nullptr);
}
